// _LiftingLayer_39994735460756
// MI455X (gfx1250) — matrix operand order repaired and checked by reading, not compiled, not run on hardware
//
#include <hip/hip_runtime.h>
#include <hip/hip_bf16.h>
#include <math.h>

typedef __attribute__((ext_vector_type(16))) _Float16 v16h;
typedef __attribute__((ext_vector_type(8)))  float    v8f;

#define NXC   16384
#define DH    128
#define KOFF  3
#define NWAVE 8

// fragment pool offsets (in halfs); each fragment = 32 lanes * 16 halfs = 512 halfs
#define WS_NODE  0        // node_w2 128x64  -> 16 frags
#define WS_EDGE  8192     // edge_w2 128x64  -> 16 frags
#define WS_COMB1 16384    // comb_w1 128x128 -> 32 frags
#define WS_COMB2 32768    // comb_w2 128x64  -> 16 frags
#define WS_TOTAL 40960

__device__ __forceinline__ float gelu_f(float v) {
    return 0.5f * v * (1.0f + erff(v * 0.70710678118654752f));
}

// LDS A-fragment swizzle: (row r, K) -> half index within a 16x128 staging region.
// Lane g*16+r holds: e<8 -> K = kb*32 + g*8 + e ; e>=8 -> K = kb*32 + 16 + g*8 + (e-8)
__device__ __forceinline__ int aswz(int r, int k) {
    int kb   = k >> 5;
    int rem  = k & 31;
    int half = rem >> 4;
    int sub  = rem & 15;
    int g    = sub >> 3;
    int wb   = sub & 7;
    int e    = half ? (8 + wb) : wb;
    return kb * 512 + (g * 16 + r) * 16 + e;
}

#define WMMA_F16(A, B, C) \
    __builtin_amdgcn_wmma_f32_16x16x32_f16(false, (A), false, (B), (short)0, (C), false, false)

// ---------------------------------------------------------------------------
// Kernel 1: swizzle layer-2 weights (f32 row-major [K][N]) into B-fragment
// layout (f16) in d_ws so the main kernel can stage them contiguously.
// B-frag layout: lane = g*16 + n, element e: K = kb*32 + (e<8 ? g*8+e : 16+g*8+(e-8)), col = nb*16 + n
// ---------------------------------------------------------------------------
__global__ void swizzle_weights(const float* __restrict__ nw2,
                                const float* __restrict__ ew2,
                                const float* __restrict__ cw1,
                                const float* __restrict__ cw2,
                                _Float16* __restrict__ ws) {
    int d = blockIdx.x * blockDim.x + threadIdx.x;
    if (d >= WS_TOTAL) return;
    const float* W; int base, ncols;
    if (d < WS_EDGE)        { W = nw2; base = WS_NODE;  ncols = 64;  }
    else if (d < WS_COMB1)  { W = ew2; base = WS_EDGE;  ncols = 64;  }
    else if (d < WS_COMB2)  { W = cw1; base = WS_COMB1; ncols = 128; }
    else                    { W = cw2; base = WS_COMB2; ncols = 64;  }
    int local = d - base;
    int f    = local >> 9;        // fragment index
    int il   = local & 511;
    int lane = il >> 4;
    int e    = il & 15;
    int NB   = ncols >> 4;
    int kb   = f / NB;
    int nb   = f % NB;
    int g    = lane >> 4;
    int n    = lane & 15;
    int Krow = kb * 32 + (e < 8 ? g * 8 + e : 16 + g * 8 + (e - 8));
    int col  = nb * 16 + n;
    ws[d] = (_Float16)W[Krow * ncols + col];
}

// ---------------------------------------------------------------------------
// Main kernel: 8 waves/block, each wave handles 4 tiles of 16 points.
// Weights live in LDS; B fragments batch-preloaded per kb step so ds_loads
// stay in flight while WMMAs execute.
// ---------------------------------------------------------------------------
__global__ __launch_bounds__(256) void lifting_kernel(
    const float* __restrict__ u0,  const float* __restrict__ x,
    const float* __restrict__ nw1, const float* __restrict__ nb1,
    const float* __restrict__ nb2v,
    const float* __restrict__ ew1, const float* __restrict__ eb1,
    const float* __restrict__ eb2v,
    const float* __restrict__ cb1, const float* __restrict__ cb2v,
    const _Float16* __restrict__ wfrag,
    float* __restrict__ out) {

    __shared__ __align__(16) _Float16 wlds[WS_TOTAL];        // 80 KB
    __shared__ __align__(32) _Float16 sA[NWAVE][2048];       // 32 KB
    __shared__ __align__(32) _Float16 sB[NWAVE][2048];       // 32 KB
    __shared__ float nbrU[NWAVE][24];
    __shared__ float nbrX[NWAVE][24];

    const int lane = threadIdx.x & 31;
    const int wv   = threadIdx.x >> 5;

    // ---- cooperative weight stage: global (pre-swizzled f16) -> LDS --------
    {
        const uint4* src = (const uint4*)wfrag;              // 81920 B = 5120 uint4
        uint4*       dst = (uint4*)wlds;
        for (int idx = threadIdx.x; idx < WS_TOTAL / 8; idx += 256)
            dst[idx] = src[idx];
    }

    // C/D layout constants: lane cg*16+cn holds rows v+8*cg, col cn (16-col tile)
    const int cg = lane >> 4;
    const int cn = lane & 15;
    // A staging write constants for h%32 == lane (see aswz derivation)
    const int subw = lane & 15;
    const int gw   = subw >> 3;
    const int ewi  = (lane >> 4) ? (8 + (subw & 7)) : (subw & 7);

    // Layer-1 weights: lane owns hidden channels h = lane + 32*hq
    float w_n0[4], w_n1[4], w_nb[4], w_e0[4], w_e1[4], w_e2[4], w_eb[4];
#pragma unroll
    for (int hq = 0; hq < 4; ++hq) {
        int h = lane + 32 * hq;
        w_n0[hq] = nw1[h];          w_n1[hq] = nw1[DH + h];     w_nb[hq] = nb1[h];
        w_e0[hq] = ew1[h];          w_e1[hq] = ew1[DH + h];
        w_e2[hq] = ew1[2 * DH + h]; w_eb[hq] = eb1[h];
    }
    // Output-column biases (per-lane constants across accumulator rows)
    float b_n2[4], b_e2[4], b_c2[4], b_c1[8];
#pragma unroll
    for (int nb = 0; nb < 4; ++nb) {
        b_n2[nb] = nb2v[nb * 16 + cn];
        b_e2[nb] = eb2v[nb * 16 + cn];
        b_c2[nb] = cb2v[nb * 16 + cn];
    }
#pragma unroll
    for (int nb = 0; nb < 8; ++nb) b_c1[nb] = cb1[nb * 16 + cn];

    const v8f vzero = {0.f, 0.f, 0.f, 0.f, 0.f, 0.f, 0.f, 0.f};

    __syncthreads();   // weights staged

    for (int t = 0; t < 4; ++t) {
        const int tile = blockIdx.x * 32 + t * NWAVE + wv;
        const int m0   = tile * 16;
        const int bb   = m0 >> 14;          // nx = 2^14
        const int i0   = m0 & (NXC - 1);

        // ---- stage clipped neighbor window [i0-3, i0+18] -------------------
        if (lane < 22) {
            int ii = i0 - KOFF + lane;
            ii = ii < 0 ? 0 : (ii > NXC - 1 ? NXC - 1 : ii);
            nbrU[wv][lane] = u0[bb * NXC + ii];
            nbrX[wv][lane] = x[bb * NXC + ii];
        }
        __syncthreads();

        // ---- layer 1 (VALU + exact-erf gelu), stage as A fragments ---------
        // edge arg: ui*e0 + uj*e1 + (xj-xi)*e2 + b = fma(uj,e1, fma(xj,e2, base_r))
        // with base_r = ui*e0 - xi*e2 + b hoisted per row.
#pragma unroll
        for (int hq = 0; hq < 4; ++hq) {
            for (int r = 0; r < 16; ++r) {
                float ui = nbrU[wv][r + KOFF];
                float xi = nbrX[wv][r + KOFF];
                float nh = gelu_f(ui * w_n0[hq] + xi * w_n1[hq] + w_nb[hq]);
                float base_r = fmaf(ui, w_e0[hq], fmaf(-xi, w_e2[hq], w_eb[hq]));
                float es = 0.f;
#pragma unroll
                for (int j = 0; j < 7; ++j) {
                    float uj = nbrU[wv][r + j];
                    float xj = nbrX[wv][r + j];
                    es += gelu_f(fmaf(uj, w_e1[hq], fmaf(xj, w_e2[hq], base_r)));
                }
                int dst = hq * 512 + (gw * 16 + r) * 16 + ewi;
                sA[wv][dst] = (_Float16)nh;
                sB[wv][dst] = (_Float16)es;
            }
        }
        __syncthreads();

        // ---- WMMA: h_self = nodeH @ node_w2 ; agg = edgeHsum @ edge_w2 -----
        v8f hs[4], ag[4];
#pragma unroll
        for (int nb = 0; nb < 4; ++nb) { hs[nb] = vzero; ag[nb] = vzero; }
#pragma unroll
        for (int kb = 0; kb < 4; ++kb) {
            v16h aN = *(const v16h*)&sA[wv][kb * 512 + lane * 16];
            v16h aE = *(const v16h*)&sB[wv][kb * 512 + lane * 16];
            v16h bN[4], bE[4];
#pragma unroll
            for (int nb = 0; nb < 4; ++nb) {
                bN[nb] = *(const v16h*)&wlds[WS_NODE + (kb * 4 + nb) * 512 + lane * 16];
                bE[nb] = *(const v16h*)&wlds[WS_EDGE + (kb * 4 + nb) * 512 + lane * 16];
            }
#pragma unroll
            for (int nb = 0; nb < 4; ++nb) {
                hs[nb] = WMMA_F16(aN, bN[nb], hs[nb]);
                ag[nb] = WMMA_F16(aE, bE[nb], ag[nb]);
            }
        }
        __syncthreads();

        // ---- stage comb input: cols [0,64)=h_self+b2, [64,128)=agg+7*b2 ----
#pragma unroll
        for (int nb = 0; nb < 4; ++nb) {
#pragma unroll
            for (int v = 0; v < 8; ++v) {
                int r = v + 8 * cg;
                sA[wv][aswz(r, nb * 16 + cn)]      = (_Float16)(hs[nb][v] + b_n2[nb]);
                sA[wv][aswz(r, 64 + nb * 16 + cn)] = (_Float16)(ag[nb][v] + 7.0f * b_e2[nb]);
            }
        }
        __syncthreads();

        // ---- WMMA: comb layer 1 (128 -> 128) -------------------------------
        v8f ch[8];
#pragma unroll
        for (int nb = 0; nb < 8; ++nb) ch[nb] = vzero;
#pragma unroll
        for (int kb = 0; kb < 4; ++kb) {
            v16h a = *(const v16h*)&sA[wv][kb * 512 + lane * 16];
            v16h bW[8];
#pragma unroll
            for (int nb = 0; nb < 8; ++nb)
                bW[nb] = *(const v16h*)&wlds[WS_COMB1 + (kb * 8 + nb) * 512 + lane * 16];
#pragma unroll
            for (int nb = 0; nb < 8; ++nb)
                ch[nb] = WMMA_F16(a, bW[nb], ch[nb]);
        }
        __syncthreads();

        // ---- bias + gelu, restage as A fragments ---------------------------
#pragma unroll
        for (int nb = 0; nb < 8; ++nb) {
#pragma unroll
            for (int v = 0; v < 8; ++v) {
                int r = v + 8 * cg;
                sB[wv][aswz(r, nb * 16 + cn)] = (_Float16)gelu_f(ch[nb][v] + b_c1[nb]);
            }
        }
        __syncthreads();

        // ---- WMMA: comb layer 2 (128 -> 64) --------------------------------
        v8f oc[4];
#pragma unroll
        for (int nb = 0; nb < 4; ++nb) oc[nb] = vzero;
#pragma unroll
        for (int kb = 0; kb < 4; ++kb) {
            v16h a = *(const v16h*)&sB[wv][kb * 512 + lane * 16];
            v16h bW[4];
#pragma unroll
            for (int nb = 0; nb < 4; ++nb)
                bW[nb] = *(const v16h*)&wlds[WS_COMB2 + (kb * 4 + nb) * 512 + lane * 16];
#pragma unroll
            for (int nb = 0; nb < 4; ++nb)
                oc[nb] = WMMA_F16(a, bW[nb], oc[nb]);
        }

        // ---- bias + store output (B, nx, 64) f32 ---------------------------
#pragma unroll
        for (int nb = 0; nb < 4; ++nb) {
#pragma unroll
            for (int v = 0; v < 8; ++v) {
                int r = v + 8 * cg;
                out[(size_t)(m0 + r) * 64 + nb * 16 + cn] = oc[nb][v] + b_c2[nb];
            }
        }
        __syncthreads();
    }
}

extern "C" void kernel_launch(void* const* d_in, const int* in_sizes, int n_in,
                              void* d_out, int out_size, void* d_ws, size_t ws_size,
                              hipStream_t stream) {
    const float* u0  = (const float*)d_in[0];
    const float* xx  = (const float*)d_in[1];
    const float* nw1 = (const float*)d_in[2];
    const float* nb1 = (const float*)d_in[3];
    const float* nw2 = (const float*)d_in[4];
    const float* nb2 = (const float*)d_in[5];
    const float* ew1 = (const float*)d_in[6];
    const float* eb1 = (const float*)d_in[7];
    const float* ew2 = (const float*)d_in[8];
    const float* eb2 = (const float*)d_in[9];
    const float* cw1 = (const float*)d_in[10];
    const float* cb1 = (const float*)d_in[11];
    const float* cw2 = (const float*)d_in[12];
    const float* cb2 = (const float*)d_in[13];

    _Float16* ws = (_Float16*)d_ws;
    float* out   = (float*)d_out;

    swizzle_weights<<<(WS_TOTAL + 255) / 256, 256, 0, stream>>>(nw2, ew2, cw1, cw2, ws);

    // 262144 points / 16 rows per tile = 16384 tiles; 32 tiles per block
    lifting_kernel<<<512, 256, 0, stream>>>(u0, xx, nw1, nb1, nb2, ew1, eb1, eb2,
                                            cb1, cb2, ws, out);
}
